// LSTMSeq2Seq_61675730370834
// MI455X (gfx1250) — compile-verified, matrix operand order checked by reading, not run on hardware
//
#include <hip/hip_runtime.h>

#define BATCH 1024
#define SEQL  168
#define TDEC  24
#define DIN   256
#define DOUT  64
#define HID   512
#define G4H   2048

typedef __attribute__((ext_vector_type(8)))  __bf16 v8bf;
typedef __attribute__((ext_vector_type(16))) __bf16 v16bf;
typedef __attribute__((ext_vector_type(8)))  float  v8f;
typedef __attribute__((ext_vector_type(4)))  float  v4f;

// ---- WMMA fragment loaders -------------------------------------------------
// 16x16x32 bf16 A-fragment layout (ISA 7.12.2): lanes 0-15 hold M=lane,
// K{0..7,16..23}; lanes 16-31 hold M=lane-16, K{8..15,24..31}.  For a
// row-major source this is two contiguous 16B loads per lane.  The B
// fragment (32x16, K x N) mirrors this with N in place of M, so a row-major
// weight matrix W[N][K] uses the identical pattern.
__device__ __forceinline__ v16bf frag_bf16(const __bf16* __restrict__ base,
                                           int row0, int lda, int kbase, int lane) {
  const int m    = lane & 15;
  const int koff = (lane >> 4) << 3;
  const __bf16* p = base + (size_t)(row0 + m) * lda + kbase + koff;
  v8bf lo = *(const v8bf*)p;
  v8bf hi = *(const v8bf*)(p + 16);
  v16bf r;
#pragma unroll
  for (int i = 0; i < 8; ++i) { r[i] = lo[i]; r[i + 8] = hi[i]; }
  return r;
}

__device__ __forceinline__ v16bf frag_f32(const float* __restrict__ base,
                                          int row0, int lda, int kbase, int lane) {
  const int m    = lane & 15;
  const int koff = (lane >> 4) << 3;
  const float* p = base + (size_t)(row0 + m) * lda + kbase + koff;
  v4f a0 = *(const v4f*)p;
  v4f a1 = *(const v4f*)(p + 4);
  v4f b0 = *(const v4f*)(p + 16);
  v4f b1 = *(const v4f*)(p + 20);
  v16bf r;
#pragma unroll
  for (int i = 0; i < 4; ++i) {
    r[i]      = (__bf16)a0[i];
    r[i + 4]  = (__bf16)a1[i];
    r[i + 8]  = (__bf16)b0[i];
    r[i + 12] = (__bf16)b1[i];
  }
  return r;
}

__device__ __forceinline__ float sigm(float x) { return 1.0f / (1.0f + __expf(-x)); }

// ---- fused LSTM step: gates = bias + X@Wih^T + Hprev@Whh^T ; update c,h ----
// One wave owns a 32(batch) x 16(hidden-unit) tile and all four gates for it:
// 8 accumulator tiles.  Each B (weight) fragment is reused across the two
// batch sub-tiles (1.5 loads/WMMA), each A fragment across the four gates.
template <bool XF32>
__global__ __launch_bounds__(256) void lstm_step_k(
    const void* __restrict__ Xin, int ldx, int Kx,
    const __bf16* __restrict__ Wih,   // [4H][Kx]  bf16 row-major
    const __bf16* __restrict__ Whh,   // [4H][HID] bf16 row-major
    const float* __restrict__ bias,   // [4H] = bih+bhh
    const __bf16* __restrict__ Hprev, // [B][HID]
    float* __restrict__ C,            // [B][HID] in-place
    __bf16* __restrict__ Hout)        // [B][HID] (must not alias Hprev)
{
  const int wave = blockIdx.x * (blockDim.x >> 5) + (threadIdx.x >> 5);
  const int lane = threadIdx.x & 31;
  const int jt = wave & 31;          // 32 column tiles over HID
  const int bt = wave >> 5;          // 32 row tiles of 32 batch rows
  const int row0 = bt << 5;
  const int jrow = jt << 4;

  v8f acc[2][4];
#pragma unroll
  for (int mb = 0; mb < 2; ++mb)
#pragma unroll
    for (int g = 0; g < 4; ++g)
#pragma unroll
      for (int i = 0; i < 8; ++i) acc[mb][g][i] = 0.0f;

  // input-to-hidden contribution
  for (int k = 0; k < Kx; k += 32) {
    v16bf a0, a1;
    if constexpr (XF32) {
      a0 = frag_f32((const float*)Xin, row0,      ldx, k, lane);
      a1 = frag_f32((const float*)Xin, row0 + 16, ldx, k, lane);
    } else {
      a0 = frag_bf16((const __bf16*)Xin, row0,      ldx, k, lane);
      a1 = frag_bf16((const __bf16*)Xin, row0 + 16, ldx, k, lane);
    }
#pragma unroll
    for (int g = 0; g < 4; ++g) {
      v16bf b = frag_bf16(Wih, g * HID + jrow, Kx, k, lane);
      acc[0][g] = __builtin_amdgcn_wmma_f32_16x16x32_bf16(
          false, a0, false, b, (short)0, acc[0][g], false, false);
      acc[1][g] = __builtin_amdgcn_wmma_f32_16x16x32_bf16(
          false, a1, false, b, (short)0, acc[1][g], false, false);
    }
  }
  // hidden-to-hidden contribution
  for (int k = 0; k < HID; k += 32) {
    v16bf a0 = frag_bf16(Hprev, row0,      HID, k, lane);
    v16bf a1 = frag_bf16(Hprev, row0 + 16, HID, k, lane);
#pragma unroll
    for (int g = 0; g < 4; ++g) {
      v16bf b = frag_bf16(Whh, g * HID + jrow, HID, k, lane);
      acc[0][g] = __builtin_amdgcn_wmma_f32_16x16x32_bf16(
          false, a0, false, b, (short)0, acc[0][g], false, false);
      acc[1][g] = __builtin_amdgcn_wmma_f32_16x16x32_bf16(
          false, a1, false, b, (short)0, acc[1][g], false, false);
    }
  }

  // fused epilogue: bias + activations + cell/hidden update
  const int n  = lane & 15;
  const int mh = (lane >> 4) << 3;
  const int j  = jrow + n;
  const float bi = bias[j];
  const float bf = bias[HID + j];
  const float bg = bias[2 * HID + j];
  const float bo = bias[3 * HID + j];
#pragma unroll
  for (int mb = 0; mb < 2; ++mb) {
#pragma unroll
    for (int r = 0; r < 8; ++r) {
      const int b = row0 + (mb << 4) + mh + r;  // C/D layout: VGPR r -> M=r (+8 hi lanes)
      float* cp = C + (size_t)b * HID + j;
      const float cold = *cp;
      const float ig = sigm(acc[mb][0][r] + bi);
      const float fg = sigm(acc[mb][1][r] + bf);
      const float gg = tanhf(acc[mb][2][r] + bg);
      const float og = sigm(acc[mb][3][r] + bo);
      const float cn = fg * cold + ig * gg;
      *cp = cn;
      Hout[(size_t)b * HID + j] = (__bf16)(og * tanhf(cn));
    }
  }
}

// ---- projection: P[B][64] = H[B][512] @ W[64][512]^T + b ------------------
__global__ __launch_bounds__(256) void proj_k(
    const __bf16* __restrict__ Hin,
    const __bf16* __restrict__ W,     // [DOUT][HID]
    const float* __restrict__ bias,   // [DOUT]
    float* __restrict__ outf, int ldo,
    __bf16* __restrict__ outb)
{
  const int wave = blockIdx.x * (blockDim.x >> 5) + (threadIdx.x >> 5);
  const int lane = threadIdx.x & 31;
  const int jt = wave & 3;    // 4 tiles over DOUT
  const int bt = wave >> 2;   // 64 tiles over BATCH
  const int row0 = bt << 4, jrow = jt << 4;

  v8f acc;
#pragma unroll
  for (int i = 0; i < 8; ++i) acc[i] = 0.0f;

  for (int k = 0; k < HID; k += 32) {
    v16bf a = frag_bf16(Hin, row0, HID, k, lane);
    v16bf b = frag_bf16(W, jrow, HID, k, lane);
    acc = __builtin_amdgcn_wmma_f32_16x16x32_bf16(
        false, a, false, b, (short)0, acc, false, false);
  }

  const int n = lane & 15, mh = (lane >> 4) << 3, j = jrow + n;
  const float bj = bias[j];
#pragma unroll
  for (int r = 0; r < 8; ++r) {
    const int b = row0 + mh + r;
    const float p = acc[r] + bj;
    if (outf) outf[(size_t)b * ldo + j] = p;
    if (outb) outb[(size_t)b * DOUT + j] = (__bf16)p;
  }
}

// ---- utility kernels -------------------------------------------------------
__global__ void cvt_bf16_k(const float* __restrict__ s, __bf16* __restrict__ d, int n) {
  int i = blockIdx.x * blockDim.x + threadIdx.x;
  if (i < n) d[i] = (__bf16)s[i];
}
__global__ void addbias_k(const float* __restrict__ a, const float* __restrict__ b,
                          float* __restrict__ o, int n) {
  int i = blockIdx.x * blockDim.x + threadIdx.x;
  if (i < n) o[i] = a[i] + b[i];
}
__global__ void zero_state_k(float* __restrict__ c0, float* __restrict__ c1,
                             __bf16* __restrict__ hz0, __bf16* __restrict__ hz1, int n) {
  int i = blockIdx.x * blockDim.x + threadIdx.x;
  if (i < n) {
    c0[i] = 0.0f; c1[i] = 0.0f;
    hz0[i] = (__bf16)0.0f; hz1[i] = (__bf16)0.0f;
  }
}

// ---- host orchestration ----------------------------------------------------
extern "C" void kernel_launch(void* const* d_in, const int* in_sizes, int n_in,
                              void* d_out, int out_size, void* d_ws, size_t ws_size,
                              hipStream_t stream) {
  (void)in_sizes; (void)n_in; (void)out_size; (void)ws_size;
  const float* x      = (const float*)d_in[0];
  const float* eWih0f = (const float*)d_in[1];
  const float* eWhh0f = (const float*)d_in[2];
  const float* eBih0  = (const float*)d_in[3];
  const float* eBhh0  = (const float*)d_in[4];
  const float* eWih1f = (const float*)d_in[5];
  const float* eWhh1f = (const float*)d_in[6];
  const float* eBih1  = (const float*)d_in[7];
  const float* eBhh1  = (const float*)d_in[8];
  const float* dWih0f = (const float*)d_in[9];
  const float* dWhh0f = (const float*)d_in[10];
  const float* dBih0  = (const float*)d_in[11];
  const float* dBhh0  = (const float*)d_in[12];
  const float* dWih1f = (const float*)d_in[13];
  const float* dWhh1f = (const float*)d_in[14];
  const float* dBih1  = (const float*)d_in[15];
  const float* dBhh1  = (const float*)d_in[16];
  const float* fcoWf  = (const float*)d_in[17];
  const float* fcoB   = (const float*)d_in[18];
  const float* fciWf  = (const float*)d_in[19];
  const float* fciB   = (const float*)d_in[20];
  float* out = (float*)d_out;

  char* ws = (char*)d_ws;
  size_t off = 0;
  auto alloc = [&](size_t bytes) -> void* {
    off = (off + 255) & ~(size_t)255;
    void* p = ws + off;
    off += bytes;
    return p;
  };

  __bf16* wih0  = (__bf16*)alloc((size_t)G4H * DIN * 2);
  __bf16* whh0  = (__bf16*)alloc((size_t)G4H * HID * 2);
  __bf16* wih1  = (__bf16*)alloc((size_t)G4H * HID * 2);
  __bf16* whh1  = (__bf16*)alloc((size_t)G4H * HID * 2);
  __bf16* dwih0 = (__bf16*)alloc((size_t)G4H * DOUT * 2);
  __bf16* dwhh0 = (__bf16*)alloc((size_t)G4H * HID * 2);
  __bf16* dwih1 = (__bf16*)alloc((size_t)G4H * HID * 2);
  __bf16* dwhh1 = (__bf16*)alloc((size_t)G4H * HID * 2);
  __bf16* fco   = (__bf16*)alloc((size_t)DOUT * HID * 2);
  __bf16* fci   = (__bf16*)alloc((size_t)DOUT * HID * 2);
  float*  be0   = (float*)alloc((size_t)G4H * 4);
  float*  be1   = (float*)alloc((size_t)G4H * 4);
  float*  bd0   = (float*)alloc((size_t)G4H * 4);
  float*  bd1   = (float*)alloc((size_t)G4H * 4);
  __bf16* h0[2] = { (__bf16*)alloc((size_t)BATCH * HID * 2),
                    (__bf16*)alloc((size_t)BATCH * HID * 2) };
  __bf16* h1[2] = { (__bf16*)alloc((size_t)BATCH * HID * 2),
                    (__bf16*)alloc((size_t)BATCH * HID * 2) };
  float*  c0    = (float*)alloc((size_t)BATCH * HID * 4);
  float*  c1    = (float*)alloc((size_t)BATCH * HID * 4);
  __bf16* decin = (__bf16*)alloc((size_t)BATCH * DOUT * 2);

  auto cvt = [&](const float* s, __bf16* d, int n) {
    cvt_bf16_k<<<(n + 255) / 256, 256, 0, stream>>>(s, d, n);
  };
  cvt(eWih0f, wih0,  G4H * DIN);
  cvt(eWhh0f, whh0,  G4H * HID);
  cvt(eWih1f, wih1,  G4H * HID);
  cvt(eWhh1f, whh1,  G4H * HID);
  cvt(dWih0f, dwih0, G4H * DOUT);
  cvt(dWhh0f, dwhh0, G4H * HID);
  cvt(dWih1f, dwih1, G4H * HID);
  cvt(dWhh1f, dwhh1, G4H * HID);
  cvt(fcoWf,  fco,   DOUT * HID);
  cvt(fciWf,  fci,   DOUT * HID);

  addbias_k<<<(G4H + 255) / 256, 256, 0, stream>>>(eBih0, eBhh0, be0, G4H);
  addbias_k<<<(G4H + 255) / 256, 256, 0, stream>>>(eBih1, eBhh1, be1, G4H);
  addbias_k<<<(G4H + 255) / 256, 256, 0, stream>>>(dBih0, dBhh0, bd0, G4H);
  addbias_k<<<(G4H + 255) / 256, 256, 0, stream>>>(dBih1, dBhh1, bd1, G4H);

  const int NH = BATCH * HID;
  zero_state_k<<<(NH + 255) / 256, 256, 0, stream>>>(c0, c1, h0[0], h1[0], NH);

  const int STEP_BLOCKS = (BATCH / 32) * (HID / 16) / 8;  // 128 blocks (8 waves ea)
  const int PROJ_BLOCKS = (BATCH / 16) * (DOUT / 16) / 8; // 32 blocks

  int p0 = 0, p1 = 0;
  // ---------------- encoder: layer0 and layer1 interleaved per step --------
  for (int t = 0; t < SEQL; ++t) {
    lstm_step_k<true><<<STEP_BLOCKS, 256, 0, stream>>>(
        (const void*)(x + (size_t)t * DIN), SEQL * DIN, DIN,
        wih0, whh0, be0, h0[p0], c0, h0[p0 ^ 1]);
    p0 ^= 1;
    lstm_step_k<false><<<STEP_BLOCKS, 256, 0, stream>>>(
        (const void*)h0[p0], HID, HID,
        wih1, whh1, be1, h1[p1], c1, h1[p1 ^ 1]);
    p1 ^= 1;
  }
  // ---------------- initial decoder input: fc_init(h1_final) ---------------
  proj_k<<<PROJ_BLOCKS, 256, 0, stream>>>(h1[p1], fci, fciB,
                                          (float*)nullptr, 0, decin);
  // ---------------- decoder: 24 autoregressive steps -----------------------
  for (int t = 0; t < TDEC; ++t) {
    lstm_step_k<false><<<STEP_BLOCKS, 256, 0, stream>>>(
        (const void*)decin, DOUT, DOUT,
        dwih0, dwhh0, bd0, h0[p0], c0, h0[p0 ^ 1]);
    p0 ^= 1;
    lstm_step_k<false><<<STEP_BLOCKS, 256, 0, stream>>>(
        (const void*)h0[p0], HID, HID,
        dwih1, dwhh1, bd1, h1[p1], c1, h1[p1 ^ 1]);
    p1 ^= 1;
    proj_k<<<PROJ_BLOCKS, 256, 0, stream>>>(h1[p1], fco, fcoB,
                                            out + (size_t)t * DOUT, TDEC * DOUT,
                                            decin);
  }
}
